// OptimizedQuantizedLinear_65420941852801
// MI455X (gfx1250) — matrix operand order repaired and checked by reading, not compiled, not run on hardware
//
#include <hip/hip_runtime.h>

// Problem constants (match reference)
#define OUT_DIM   8192
#define IN_DIM    8192
#define GROUP_SZ  128
#define BATCH_SZ  64
#define KSTEPS    (IN_DIM / 32)    // 256 K-steps of 32
#define MTILES    (BATCH_SZ / 16)  // 4 M tiles of 16
#define AFRAG_H   16               // halves per lane per A fragment
#define ASTRIDE   (KSTEPS * 32 * AFRAG_H)  // halves per M-tile plane in xf

typedef __attribute__((ext_vector_type(16))) _Float16     v16h;
typedef __attribute__((ext_vector_type(2)))  _Float16     h2;
typedef __attribute__((ext_vector_type(8)))  float        v8f;
typedef __attribute__((ext_vector_type(4)))  int          v4i;
typedef __attribute__((ext_vector_type(4)))  float        v4f;
typedef __attribute__((ext_vector_type(8)))  unsigned int v8u;

// Explicit global-address-space pointers: force global_load_b128 (LOADcnt
// only, partial waits) instead of flat_load_b128 (LOADcnt+DScnt full drains).
typedef const __attribute__((address_space(1))) v16h*  gv16h_p;
typedef const __attribute__((address_space(1))) v4i*   gv4i_p;
typedef const __attribute__((address_space(1))) v4f*   gv4f_p;
typedef const __attribute__((address_space(1))) float* gf_p;
typedef __attribute__((address_space(1))) float*       gf_wp;
typedef const __attribute__((address_space(1))) int*   gi_p;

// ---------------------------------------------------------------------------
// Dequant 16 int4 codes (8 bytes in two v4i) to a B fragment via the f16
// magic number: h = 0x6400|q encodes 1024+q, so (q-8)*s = h*s + (-1032*s)
// -> 3 bit-ops + one v_pk_fma_f16 per nibble pair.
// ---------------------------------------------------------------------------
__device__ __forceinline__ v16h dequant16(v4i p0, v4i p1, h2 s2, h2 c2) {
  const unsigned MAGIC = 0x64006400u;
  v8u bw;
#pragma unroll
  for (int j = 0; j < 4; ++j) {
    const unsigned v = (unsigned)p0[j];
    const unsigned hv = (v & 0xFu) | ((v << 12) & 0xF0000u) | MAGIC;
    h2 w0 = __builtin_bit_cast(h2, hv) * s2 + c2;
    bw[j] = __builtin_bit_cast(unsigned, w0);

    const unsigned u = (unsigned)p1[j];
    const unsigned hu = (u & 0xFu) | ((u << 12) & 0xF0000u) | MAGIC;
    h2 w1 = __builtin_bit_cast(h2, hu) * s2 + c2;
    bw[4 + j] = __builtin_bit_cast(unsigned, w1);
  }
  return __builtin_bit_cast(v16h, bw);
}

__device__ __forceinline__ void wmma4(v8f acc[MTILES], const v16h& a0,
                                      const v16h& a1, const v16h& a2,
                                      const v16h& a3, const v16h& b) {
  acc[0] = __builtin_amdgcn_wmma_f32_16x16x32_f16(false, a0, false, b,
                                                  (short)0, acc[0], false, false);
  acc[1] = __builtin_amdgcn_wmma_f32_16x16x32_f16(false, a1, false, b,
                                                  (short)0, acc[1], false, false);
  acc[2] = __builtin_amdgcn_wmma_f32_16x16x32_f16(false, a2, false, b,
                                                  (short)0, acc[2], false, false);
  acc[3] = __builtin_amdgcn_wmma_f32_16x16x32_f16(false, a3, false, b,
                                                  (short)0, acc[3], false, false);
}

// ---------------------------------------------------------------------------
// Prepack x (fp32, [64, 8192]) into WMMA A-fragment layout as f16 in d_ws.
// A 16x32 f16 layout (ISA 7.12.2): lanes 0-15 -> M=lane, K {0..7,16..23};
// lanes 16-31 -> M=lane-16, K {8..15,24..31}. One thread = one lane of one
// fragment = 16 f16 = 32 contiguous bytes.
// ---------------------------------------------------------------------------
__global__ __launch_bounds__(256) void prepack_x_kernel(
    const float* __restrict__ x, _Float16* __restrict__ xf)
{
  const int tid   = blockIdx.x * 256 + threadIdx.x;  // 0 .. MTILES*KSTEPS*32-1
  const int lane  = tid & 31;
  const int ks    = (tid >> 5) & (KSTEPS - 1);
  const int mtile = tid >> 13;                       // / (32*KSTEPS)
  const int row   = mtile * 16 + (lane & 15);
  const int kbase = ks * 32 + ((lane >> 4) << 3);    // +8 for lanes 16..31

  gv4f_p src = (gv4f_p)(x + (size_t)row * IN_DIM + kbase);
  v4f f0 = src[0];
  v4f f1 = src[1];
  v4f f2 = src[4];
  v4f f3 = src[5];

  v16h a;
#pragma unroll
  for (int j = 0; j < 4; ++j) {
    a[j]      = (_Float16)f0[j];
    a[4 + j]  = (_Float16)f1[j];
    a[8 + j]  = (_Float16)f2[j];
    a[12 + j] = (_Float16)f3[j];
  }
  *(__attribute__((address_space(1))) v16h*)(xf + (size_t)tid * AFRAG_H) = a;
}

// ---------------------------------------------------------------------------
// Main kernel (prepacked-A path), software-pipelined by one K-step.
// Each wave owns 16 output columns (= 16 weight rows) and all 4 M tiles;
// 8 waves per block -> 128 columns. grid.y = K-split chunk.
// B 32x16 f16 fragment: lanes 0-15 hold K 0..7,16..23 (col N=lane%16), lanes 16-31
// hold K 8..15,24..31 -> each lane reads 8 packed bytes = 8 int32 as two 16 B loads.
// Last group / last step are peeled so no load ever goes past the buffers.
// ---------------------------------------------------------------------------
__global__ __launch_bounds__(256) void qlinear_wmma_xf_kernel(
    const _Float16* __restrict__ xf,
    const int*      __restrict__ packed, // [OUT, IN/2] one byte per int32
    const float*    __restrict__ scales, // [OUT, IN/GROUP]
    float*          __restrict__ dst,    // out (split==1) or partials base
    int ks_per_chunk)
{
  const int lane   = threadIdx.x & 31;
  const int wave   = threadIdx.x >> 5;
  const int ncol16 = lane & 15;
  const int khalf  = lane >> 4;
  const int n_base = blockIdx.x * 128 + wave * 16;
  const int o      = n_base + ncol16;           // weight row / output column
  const int chunk  = blockIdx.y;
  const int ks_begin = chunk * ks_per_chunk;

  gv4i_p pw = (gv4i_p)(packed + (size_t)o * (IN_DIM / 2)
                       + (((size_t)ks_begin * 32 + (khalf << 3)) >> 1));
  gi_p   pw_pf = (gi_p)pw;
  gf_p   sp = (gf_p)(scales + (size_t)o * (IN_DIM / GROUP_SZ) + (ks_begin >> 2));

  const size_t abase = ((size_t)ks_begin * 32 + lane) * AFRAG_H;
  gv16h_p ap0 = (gv16h_p)(xf + abase + 0 * (size_t)ASTRIDE);
  gv16h_p ap1 = (gv16h_p)(xf + abase + 1 * (size_t)ASTRIDE);
  gv16h_p ap2 = (gv16h_p)(xf + abase + 2 * (size_t)ASTRIDE);
  gv16h_p ap3 = (gv16h_p)(xf + abase + 3 * (size_t)ASTRIDE);

  v8f acc[MTILES];
#pragma unroll
  for (int m = 0; m < MTILES; ++m)
#pragma unroll
    for (int r = 0; r < 8; ++r) acc[m][r] = 0.0f;

  // ---- Pipeline prologue: buffer K-step 0.
  v4i  p0 = pw[0], p1 = pw[2]; pw += 4;
  v16h a0 = *ap0; ap0 += 32;
  v16h a1 = *ap1; ap1 += 32;
  v16h a2 = *ap2; ap2 += 32;
  v16h a3 = *ap3; ap3 += 32;

  const int ngroups = ks_per_chunk >> 2;        // 4 K-steps per scale group

  for (int g = 0; g < ngroups - 1; ++g) {
    const float s = *sp++;
    const _Float16 sh = (_Float16)s;
    const _Float16 bh = (_Float16)(-1032.0f * s);
    h2 s2; s2[0] = sh; s2[1] = sh;
    h2 c2; c2[0] = bh; c2[1] = bh;

    __builtin_prefetch((const void*)(pw_pf + 1024), 0, 1);  // 4KB ahead
    pw_pf += 64;

#pragma unroll
    for (int t = 0; t < 4; ++t) {
      // Issue next K-step's loads before consuming the current buffers.
      v4i  np0 = pw[0], np1 = pw[2]; pw += 4;
      v16h na0 = *ap0; ap0 += 32;
      v16h na1 = *ap1; ap1 += 32;
      v16h na2 = *ap2; ap2 += 32;
      v16h na3 = *ap3; ap3 += 32;

      const v16h b = dequant16(p0, p1, s2, c2);
      wmma4(acc, a0, a1, a2, a3, b);

      p0 = np0; p1 = np1;
      a0 = na0; a1 = na1; a2 = na2; a3 = na3;
    }
  }

  // ---- Last group: 3 pipelined steps + 1 epilogue step (no next load).
  {
    const float s = *sp;
    const _Float16 sh = (_Float16)s;
    const _Float16 bh = (_Float16)(-1032.0f * s);
    h2 s2; s2[0] = sh; s2[1] = sh;
    h2 c2; c2[0] = bh; c2[1] = bh;

#pragma unroll
    for (int t = 0; t < 3; ++t) {
      v4i  np0 = pw[0], np1 = pw[2]; pw += 4;
      v16h na0 = *ap0; ap0 += 32;
      v16h na1 = *ap1; ap1 += 32;
      v16h na2 = *ap2; ap2 += 32;
      v16h na3 = *ap3; ap3 += 32;

      const v16h b = dequant16(p0, p1, s2, c2);
      wmma4(acc, a0, a1, a2, a3, b);

      p0 = np0; p1 = np1;
      a0 = na0; a1 = na1; a2 = na2; a3 = na3;
    }
    const v16h b = dequant16(p0, p1, s2, c2);
    wmma4(acc, a0, a1, a2, a3, b);
  }

  // D layout: VGPR r -> lanes 0-15: M=r, N=lane; lanes 16-31: M=8+r, N=lane-16.
  gf_wp obase = (gf_wp)(dst + (size_t)chunk * BATCH_SZ * OUT_DIM
                        + (size_t)(khalf * 8) * OUT_DIM + n_base + ncol16);
#pragma unroll
  for (int m = 0; m < MTILES; ++m)
#pragma unroll
    for (int r = 0; r < 8; ++r)
      obase[(size_t)(m * 16 + r) * OUT_DIM] = acc[m][r];
}

// ---------------------------------------------------------------------------
// Fallback kernel: builds A fragments from raw fp32 x (used only when d_ws
// is too small for the prepack buffer). Not pipelined.
// ---------------------------------------------------------------------------
__global__ __launch_bounds__(256) void qlinear_wmma_raw_kernel(
    const float* __restrict__ x,
    const int*   __restrict__ packed,
    const float* __restrict__ scales,
    float*       __restrict__ dst,
    int ks_per_chunk)
{
  const int lane   = threadIdx.x & 31;
  const int wave   = threadIdx.x >> 5;
  const int ncol16 = lane & 15;
  const int khalf  = lane >> 4;
  const int n_base = blockIdx.x * 128 + wave * 16;
  const int o      = n_base + ncol16;
  const int chunk  = blockIdx.y;
  const int ks_begin = chunk * ks_per_chunk;

  gv4i_p pw = (gv4i_p)(packed + (size_t)o * (IN_DIM / 2)
                       + (((size_t)ks_begin * 32 + (khalf << 3)) >> 1));
  gf_p   sp = (gf_p)(scales + (size_t)o * (IN_DIM / GROUP_SZ) + (ks_begin >> 2));

  const size_t xbase = (size_t)ks_begin * 32 + (khalf << 3);
  gv4f_p xr0 = (gv4f_p)(x + (size_t)(0 * 16 + ncol16) * IN_DIM + xbase);
  gv4f_p xr1 = (gv4f_p)(x + (size_t)(1 * 16 + ncol16) * IN_DIM + xbase);
  gv4f_p xr2 = (gv4f_p)(x + (size_t)(2 * 16 + ncol16) * IN_DIM + xbase);
  gv4f_p xr3 = (gv4f_p)(x + (size_t)(3 * 16 + ncol16) * IN_DIM + xbase);

  v8f acc[MTILES];
#pragma unroll
  for (int m = 0; m < MTILES; ++m)
#pragma unroll
    for (int r = 0; r < 8; ++r) acc[m][r] = 0.0f;

  const int ngroups = ks_per_chunk >> 2;

  for (int g = 0; g < ngroups; ++g) {
    const float s = *sp++;
    const _Float16 sh = (_Float16)s;
    const _Float16 bh = (_Float16)(-1032.0f * s);
    h2 s2; s2[0] = sh; s2[1] = sh;
    h2 c2; c2[0] = bh; c2[1] = bh;

#pragma unroll
    for (int t = 0; t < 4; ++t) {
      const v4i p0 = pw[0], p1 = pw[2]; pw += 4;
      v4f f00 = xr0[0], f01 = xr0[1], f02 = xr0[4], f03 = xr0[5]; xr0 += 8;
      v4f f10 = xr1[0], f11 = xr1[1], f12 = xr1[4], f13 = xr1[5]; xr1 += 8;
      v4f f20 = xr2[0], f21 = xr2[1], f22 = xr2[4], f23 = xr2[5]; xr2 += 8;
      v4f f30 = xr3[0], f31 = xr3[1], f32 = xr3[4], f33 = xr3[5]; xr3 += 8;

      const v16h b = dequant16(p0, p1, s2, c2);

      v16h a0, a1, a2, a3;
#pragma unroll
      for (int j = 0; j < 4; ++j) {
        a0[j] = (_Float16)f00[j]; a0[4+j] = (_Float16)f01[j];
        a0[8+j] = (_Float16)f02[j]; a0[12+j] = (_Float16)f03[j];
        a1[j] = (_Float16)f10[j]; a1[4+j] = (_Float16)f11[j];
        a1[8+j] = (_Float16)f12[j]; a1[12+j] = (_Float16)f13[j];
        a2[j] = (_Float16)f20[j]; a2[4+j] = (_Float16)f21[j];
        a2[8+j] = (_Float16)f22[j]; a2[12+j] = (_Float16)f23[j];
        a3[j] = (_Float16)f30[j]; a3[4+j] = (_Float16)f31[j];
        a3[8+j] = (_Float16)f32[j]; a3[12+j] = (_Float16)f33[j];
      }
      wmma4(acc, a0, a1, a2, a3, b);
    }
  }

  gf_wp obase = (gf_wp)(dst + (size_t)chunk * BATCH_SZ * OUT_DIM
                        + (size_t)(khalf * 8) * OUT_DIM + n_base + ncol16);
#pragma unroll
  for (int m = 0; m < MTILES; ++m)
#pragma unroll
    for (int r = 0; r < 8; ++r)
      obase[(size_t)(m * 16 + r) * OUT_DIM] = acc[m][r];
}

// ---------------------------------------------------------------------------
// Deterministic K-split reduction: out = sum of nsplit partial matrices.
// ---------------------------------------------------------------------------
__global__ __launch_bounds__(256) void reduce_partials_kernel(
    const float* __restrict__ part, float* __restrict__ out, int nsplit)
{
  const int i = blockIdx.x * 256 + threadIdx.x;
  const size_t n = (size_t)BATCH_SZ * OUT_DIM;
  gf_p p = (gf_p)(part + i);
  float s = 0.0f;
  for (int c = 0; c < nsplit; ++c) s += p[(size_t)c * n];
  *(gf_wp)(out + i) = s;
}

extern "C" void kernel_launch(void* const* d_in, const int* in_sizes, int n_in,
                              void* d_out, int out_size, void* d_ws, size_t ws_size,
                              hipStream_t stream) {
  const float* x      = (const float*)d_in[0];
  const int*   packed = (const int*)d_in[1];
  const float* scales = (const float*)d_in[2];
  float*       out    = (float*)d_out;

  const size_t xf_bytes  = (size_t)MTILES * KSTEPS * 32 * AFRAG_H * sizeof(_Float16); // 1 MB
  const size_t out_bytes = (size_t)BATCH_SZ * OUT_DIM * sizeof(float);                // 2 MB

  _Float16* xf   = (_Float16*)d_ws;
  float* partial = (float*)((char*)d_ws + xf_bytes);

  int use_xf, split;
  if (ws_size >= xf_bytes + 8 * out_bytes)      { use_xf = 1; split = 8; }
  else if (ws_size >= xf_bytes + 4 * out_bytes) { use_xf = 1; split = 4; }
  else if (ws_size >= xf_bytes)                 { use_xf = 1; split = 1; }
  else                                          { use_xf = 0; split = 1; }

  if (use_xf) {
    prepack_x_kernel<<<(MTILES * KSTEPS * 32) / 256, 256, 0, stream>>>(x, xf);
    qlinear_wmma_xf_kernel<<<dim3(OUT_DIM / 128, split), 256, 0, stream>>>(
        xf, packed, scales, (split > 1) ? partial : out, KSTEPS / split);
  } else {
    qlinear_wmma_raw_kernel<<<dim3(OUT_DIM / 128, split), 256, 0, stream>>>(
        x, packed, scales, (split > 1) ? partial : out, KSTEPS / split);
  }

  if (split > 1) {
    reduce_partials_kernel<<<(BATCH_SZ * OUT_DIM) / 256, 256, 0, stream>>>(
        partial, out, split);
  }
}
